// FP8Linear_76330158784784
// MI455X (gfx1250) — 8-bit matrix operand order repaired and checked by reading, compiled, not run on hardware
//
#include <hip/hip_runtime.h>
#include <hip/hip_bf16.h>

typedef __attribute__((ext_vector_type(16))) int   v16i;
typedef __attribute__((ext_vector_type(8)))  float v8f;
typedef __attribute__((ext_vector_type(4)))  int   v4i;

typedef __attribute__((address_space(1))) v4i* g_v4i_p;
typedef __attribute__((address_space(3))) v4i* l_v4i_p;

#if defined(__has_builtin)
#  if __has_builtin(__builtin_amdgcn_global_load_async_to_lds_b128)
#    define HAVE_ASYNC_LDS 1
#  else
#    define HAVE_ASYNC_LDS 0
#  endif
#  if __has_builtin(__builtin_amdgcn_s_wait_asynccnt)
#    define WAIT_ASYNC(n) __builtin_amdgcn_s_wait_asynccnt(n)
#  else
#    define WAIT_ASYNC(n) asm volatile("s_wait_asynccnt %0" ::"i"(n) : "memory")
#  endif
#else
#  define HAVE_ASYNC_LDS 0
#  define WAIT_ASYNC(n)
#endif

#define E4M3_MAX 448.0f

__device__ __forceinline__ unsigned int f32_to_e4m3(float f) {
    unsigned int sgn = (__float_as_uint(f) >> 31) << 7;
    float a = fminf(fabsf(f), E4M3_MAX);
    unsigned int code;
    if (a < 0.015625f) {
        code = (unsigned int)rintf(a * 512.0f);
    } else {
        int e;
        (void)frexpf(a, &e);
        int ex = e - 1;
        unsigned int qi = (unsigned int)rintf(ldexpf(a, 3 - ex));
        if (qi == 16u) { qi = 8u; ex += 1; }
        code = ((unsigned int)(ex + 7) << 3) | (qi - 8u);
    }
    return sgn | code;
}

__global__ void __launch_bounds__(256) quant_e4m3_kernel(
    const float* __restrict__ in, unsigned int* __restrict__ outq,
    float* __restrict__ scales, int nblocks, int kbl, int nrows)
{
    const int lane = threadIdx.x & 31;
    const int blk  = blockIdx.x * (blockDim.x >> 5) + (threadIdx.x >> 5);
    if (blk >= nblocks) return;

    const float4 v = *(const float4*)(in + (size_t)blk * 128 + lane * 4);
    float am = fmaxf(fmaxf(fabsf(v.x), fabsf(v.y)), fmaxf(fabsf(v.z), fabsf(v.w)));
#pragma unroll
    for (int off = 16; off > 0; off >>= 1)
        am = fmaxf(am, __shfl_xor(am, off, 32));
    am = fmaxf(am, 1e-10f);

    const float scale = am * (1.0f / E4M3_MAX);
    const float inv   = E4M3_MAX / am;

    unsigned int p =  f32_to_e4m3(v.x * inv)
                   | (f32_to_e4m3(v.y * inv) << 8)
                   | (f32_to_e4m3(v.z * inv) << 16)
                   | (f32_to_e4m3(v.w * inv) << 24);
    outq[(size_t)blk * 32 + lane] = p;
    if (lane == 0) {
        const int row = blk / kbl;
        const int kb  = blk - row * kbl;
        scales[(size_t)kb * nrows + row] = scale;
    }
}

#define TILE_N     64
#define LDS_PITCH  144
#define BUF_BYTES  (TILE_N * LDS_PITCH)

__global__ void __launch_bounds__(256) fp8_gemm_kernel(
    const unsigned char* __restrict__ xq,
    const float*         __restrict__ xs,
    const unsigned char* __restrict__ wq,
    const float*         __restrict__ wsc,
    const float*         __restrict__ bias,
    float*               __restrict__ out,
    int M, int N, int K)
{
    __shared__ __align__(16) unsigned char sB[2 * BUF_BYTES];

    const int KB   = K >> 7;
    const int tid  = threadIdx.x;
    const int lane = tid & 31;
    const int wid  = tid >> 5;
    const int ln   = lane & 15;
    const int hl   = lane >> 4;

    const int nst = N >> 6;
    const int n0  = (blockIdx.x % nst) << 6;
    const int m0  = ((blockIdx.x / nst) << 7) + (wid << 4);
    const int m0u = __builtin_amdgcn_readfirstlane(m0);

    auto stage = [&](int kb, int buf) {
#pragma unroll
        for (int i = 0; i < 2; ++i) {
            const int c   = tid + (i << 8);
            const int row = c >> 3;
            const int col = c & 7;
            const unsigned char* g = wq + (size_t)(n0 + row) * K + (size_t)kb * 128 + col * 16;
            unsigned char*       l = sB + buf * BUF_BYTES + row * LDS_PITCH + col * 16;
#if HAVE_ASYNC_LDS
            __builtin_amdgcn_global_load_async_to_lds_b128(
                (g_v4i_p)g, (l_v4i_p)l, 0, 0);
#else
            *(uint4*)l = *(const uint4*)g;
#endif
        }
    };

    v8f acc[4] = {v8f{}, v8f{}, v8f{}, v8f{}};

    stage(0, 0);
    if (KB > 1) stage(1, 1);

    const unsigned char* aRow = xq + (size_t)(m0 + ln) * K + hl * 16;

    for (int kb = 0; kb < KB; ++kb) {
        const int buf = kb & 1;
#if HAVE_ASYNC_LDS
        if (kb + 1 < KB) { WAIT_ASYNC(2); }
        else             { WAIT_ASYNC(0); }
#endif
        __syncthreads();

        v16i a;
        const unsigned char* ak = aRow + (size_t)kb * 128;
#pragma unroll
        for (int h = 0; h < 2; ++h) {
#pragma unroll
            for (int g = 0; g < 4; ++g) {
                const uint2 d = *(const uint2*)(ak + h * 64 + (g >> 1) * 32 + (g & 1) * 8);
                a[h * 8 + 2 * g]     = (int)d.x;
                a[h * 8 + 2 * g + 1] = (int)d.y;
            }
        }
        if (kb + 2 < KB) __builtin_prefetch(ak + 256, 0, 0);

        const float4* sxp = (const float4*)(xs + (size_t)kb * M + m0u);
        const float4 s0 = sxp[0], s1 = sxp[1], s2 = sxp[2], s3 = sxp[3];
        float sx[8];
        sx[0] = hl ? s2.x : s0.x;  sx[1] = hl ? s2.y : s0.y;
        sx[2] = hl ? s2.z : s0.z;  sx[3] = hl ? s2.w : s0.w;
        sx[4] = hl ? s3.x : s1.x;  sx[5] = hl ? s3.y : s1.y;
        sx[6] = hl ? s3.z : s1.z;  sx[7] = hl ? s3.w : s1.w;

        const float* swRow = wsc + (size_t)kb * N + n0 + ln;

#define DO_TILE(T, RA)                                                          \
        {                                                                       \
            const unsigned char* bk =                                           \
                sB + buf * BUF_BYTES + ((T) * 16 + ln) * LDS_PITCH + hl * 16;   \
            v16i b;                                                             \
            _Pragma("unroll")                                                   \
            for (int g = 0; g < 4; ++g) {                                       \
                const uint4 d = *(const uint4*)(bk + g * 32);                   \
                b[4 * g]     = (int)d.x;                                        \
                b[4 * g + 1] = (int)d.y;                                        \
                b[4 * g + 2] = (int)d.z;                                        \
                b[4 * g + 3] = (int)d.w;                                        \
            }                                                                   \
            v8f c = {};                                                         \
            c = __builtin_amdgcn_wmma_f32_16x16x128_fp8_fp8(a, b, (short)0, c,  \
                                                            RA, false);        \
            const float sw = swRow[(T) * 16];                                   \
            _Pragma("unroll")                                                   \
            for (int r = 0; r < 8; ++r)                                         \
                acc[T][r] += c[r] * (sx[r] * sw);                               \
        }

        DO_TILE(0, false)
        DO_TILE(1, true)
        DO_TILE(2, true)
        DO_TILE(3, true)
#undef DO_TILE

        __syncthreads();
        if (kb + 2 < KB) stage(kb + 2, buf);
    }

#pragma unroll
    for (int t = 0; t < 4; ++t) {
        const int n  = n0 + t * 16 + ln;
        const float bv = bias[n];
#pragma unroll
        for (int r = 0; r < 8; ++r) {
            const int m = m0 + r + 8 * hl;
            out[(size_t)m * N + n] = acc[t][r] + bv;
        }
    }
}

extern "C" void kernel_launch(void* const* d_in, const int* in_sizes, int n_in,
                              void* d_out, int out_size, void* d_ws, size_t ws_size,
                              hipStream_t stream) {
    const float* x    = (const float*)d_in[0];
    const float* w    = (const float*)d_in[1];
    const float* bias = (const float*)d_in[2];
    float* out = (float*)d_out;

    const int N = in_sizes[2];
    const int K = in_sizes[1] / N;
    const int M = in_sizes[0] / K;
    const int KB = K >> 7;

    unsigned char* ws8 = (unsigned char*)d_ws;
    unsigned int* xq  = (unsigned int*)(ws8);
    unsigned int* wq  = (unsigned int*)(ws8 + (size_t)M * K);
    float*        xsc = (float*)(ws8 + (size_t)M * K + (size_t)N * K);
    float*        wsc = (float*)(ws8 + (size_t)M * K + (size_t)N * K + (size_t)M * KB * 4);

    const int xblocks = (M * K) >> 7;
    const int wblocks = (N * K) >> 7;
    const int wpb = 256 / 32;

    quant_e4m3_kernel<<<(xblocks + wpb - 1) / wpb, 256, 0, stream>>>(x, xq, xsc, xblocks, KB, M);
    quant_e4m3_kernel<<<(wblocks + wpb - 1) / wpb, 256, 0, stream>>>(w, wq, wsc, wblocks, KB, N);

    const int blocks = (M >> 7) * (N >> 6);
    fp8_gemm_kernel<<<blocks, 256, 0, stream>>>(
        (const unsigned char*)xq, xsc, (const unsigned char*)wq, wsc, bias, out, M, N, K);
}
